// WindowAttentionBlock_60387240182509
// MI455X (gfx1250) — compile-verified, matrix operand order checked by reading, not run on hardware
//
#include <hip/hip_runtime.h>
#include <stdint.h>

#define NUM_HEADS 8
#define KEY_DIM   64
#define SHIFT     4
#define CC        512    // channels
#define T         64     // tokens per 8x8 window
#define NWIN      512    // 8 batches * 8 * 8 windows

#define XSTR 520         // padded LDS strides (bank-conflict-free column access)
#define HSTR 72
#define SSTR 68

#define WMAT (512 * 512) // elements per weight matrix

typedef __bf16 bf16_t;
typedef bf16_t v16bf __attribute__((ext_vector_type(16)));
typedef bf16_t v4bf  __attribute__((ext_vector_type(4)));
typedef float  v8f   __attribute__((ext_vector_type(8)));

// hardware f32 -> bf16 (RNE); lowers to v_cvt_pk_bf16_f32 on gfx1250
__device__ __forceinline__ bf16_t f2bf(float f) { return (bf16_t)f; }

union FragU { uint4 q[2]; bf16_t h[16]; v16bf v; };

// 16x32 bf16 fragment from a row-major [rows][stride] bf16 buffer (LDS or global).
// Used as A (rows = M) or as B when the buffer holds B^T (rows = N).
// Per ISA layout: lanes 0-15 hold K {0..7,16..23}, lanes 16-31 hold K {8..15,24..31}.
__device__ __forceinline__ v16bf frag_nk(const bf16_t* buf, int stride,
                                         int rowBase, int kBase, int lane) {
  const int half = lane >> 4;
  const int r = rowBase + (lane & 15);
  FragU u;
  u.q[0] = *(const uint4*)(buf + r * stride + kBase + half * 8);
  u.q[1] = *(const uint4*)(buf + r * stride + kBase + 16 + half * 8);
  return u.v;
}

// Fallback B fragment (32x16) from a global f32 matrix stored [K][N], row stride S.
__device__ __forceinline__ v16bf frag_w(const float* w, int S, int kBase, int nBase, int lane) {
  const int half = lane >> 4;
  const float* base  = w + (size_t)(kBase + half * 8) * S + nBase + (lane & 15);
  const float* base2 = base + 16 * S;
  FragU u;
#pragma unroll
  for (int i = 0; i < 8; ++i) u.h[i]     = f2bf(base[i * S]);
#pragma unroll
  for (int i = 0; i < 8; ++i) u.h[8 + i] = f2bf(base2[i * S]);
  return u.v;
}

// Store a 16x16 f32 C tile into a bf16 LDS buffer (optionally transposed).
// C layout: VGPR vg holds row M = vg + 8*(lane>=16), N = lane&15.
__device__ __forceinline__ void store_c_bf16(bf16_t* buf, int stride, int mBase, int nBase,
                                             v8f c, int lane, bool transpose) {
  const int half = lane >> 4;
  const int nn = nBase + (lane & 15);
#pragma unroll
  for (int vg = 0; vg < 8; ++vg) {
    const int m = mBase + vg + 8 * half;
    if (transpose) buf[nn * stride + m] = f2bf(c[vg]);
    else           buf[m * stride + nn] = f2bf(c[vg]);
  }
}

// ---- one-time weight repack: f32 [K][N] -> bf16 [N][K] (B-transposed) in ws ----
// layout: wt[0..2][h][d][c] for Wq/Wk/Wv (stride CC), wt[3][h][c][d] for Wo (stride 64)
__global__ __launch_bounds__(256)
void prep_weights_kernel(const float* __restrict__ wq, const float* __restrict__ wk,
                         const float* __restrict__ wv, const float* __restrict__ wo,
                         bf16_t* __restrict__ wt) {
  const int gid = blockIdx.x * 256 + threadIdx.x;   // 4 * 512 * 512 threads total
  const int mat = gid >> 18;
  const int rem = gid & (WMAT - 1);
  float v;
  if (mat < 3) {
    const int c = rem & 511;
    const int hd = rem >> 9;            // h*64 + d
    const int h = hd >> 6, d = hd & 63;
    const float* w = (mat == 0) ? wq : (mat == 1) ? wk : wv;
    v = w[(c * NUM_HEADS + h) * KEY_DIM + d];
  } else {
    const int d = rem & 63;
    const int hc = rem >> 6;            // h*512 + c
    const int h = hc >> 9, c = hc & 511;
    v = wo[((size_t)h * KEY_DIM + d) * CC + c];
  }
  wt[gid] = f2bf(v);
}

template <bool PREP>
__global__ __launch_bounds__(256)
void swin_window_attn_kernel(const float* __restrict__ x,
                             const float* __restrict__ wq, const float* __restrict__ bq,
                             const float* __restrict__ wk, const float* __restrict__ bk,
                             const float* __restrict__ wv, const float* __restrict__ bv,
                             const float* __restrict__ wo, const float* __restrict__ bo,
                             const bf16_t* __restrict__ wt,
                             float* __restrict__ out) {
  __shared__ __align__(16) bf16_t Xs[T * XSTR];   // ~65 KB window activations (bf16)
  __shared__ __align__(16) bf16_t Qs[T * HSTR];   //  9 KB Q_h, then attn weights
  __shared__ __align__(16) bf16_t Ks[T * HSTR];   //  9 KB K_h [tok][d], then O_h [tok][d]
  __shared__ __align__(16) bf16_t Vs[T * HSTR];   //  9 KB V_h transposed [d][tok]
  __shared__ __align__(16) float  Ss[T * SSTR];   // 17 KB raw scores

  const int tid  = threadIdx.x;
  const int lane = tid & 31;
  const int wave = tid >> 5;

  const int nwin = blockIdx.x;        // window id
  const int b    = nwin >> 6;
  const int win  = nwin & 63;
  const int whi  = win >> 3, wwi = win & 7;

  // ---- Phase 0: gather shifted window into LDS as bf16 ----
  {
    const int t   = tid >> 2;         // 4 threads per token row
    const int sub = tid & 3;
    const int r = t >> 3, c = t & 7;
    const int gy = (whi * 8 + r + SHIFT) & 63;
    const int gx = (wwi * 8 + c + SHIFT) & 63;
    const float* src = x + ((size_t)((b * 64 + gy) * 64 + gx)) * CC + sub * 128;
    bf16_t* dst = Xs + t * XSTR + sub * 128;
#pragma unroll 4
    for (int i = 0; i < 32; ++i) {
      float4 f = ((const float4*)src)[i];
      v4bf p = { f2bf(f.x), f2bf(f.y), f2bf(f.z), f2bf(f.w) };
      *(v4bf*)(dst + i * 4) = p;
    }
  }
  __syncthreads();

  // Persistent projection accumulators: wave covers out columns [wave*64, wave*64+64)
  v8f acc[16];
#pragma unroll
  for (int i = 0; i < 16; ++i) acc[i] = (v8f){0.f,0.f,0.f,0.f,0.f,0.f,0.f,0.f};

  const float scale = 0.125f;  // 1/sqrt(KEY_DIM)

#pragma unroll 1
  for (int h = 0; h < NUM_HEADS; ++h) {
    if (PREP && h + 1 < NUM_HEADS) {
      // hint next head's weight slabs toward L2/L0 (global_prefetch_b8)
      __builtin_prefetch(wt + (size_t)(h + 1) * KEY_DIM * CC, 0, 1);
      __builtin_prefetch(wt + WMAT     + (size_t)(h + 1) * KEY_DIM * CC, 0, 1);
      __builtin_prefetch(wt + 2 * WMAT + (size_t)(h + 1) * KEY_DIM * CC, 0, 1);
      __builtin_prefetch(wt + 3 * WMAT + (size_t)(h + 1) * CC * KEY_DIM, 0, 1);
    }

    // ---- Q/K/V projections for this head: 48 16x16 tiles, 6 per wave ----
#pragma unroll 1
    for (int i = 0; i < 6; ++i) {
      const int job  = wave * 6 + i;
      const int G    = job >> 4;              // 0=Q 1=K 2=V
      const int tile = job & 15;
      const int m = tile >> 2, nn = tile & 3;
      const float* wg = (G == 0) ? wq : (G == 1) ? wk : wv;
      const float* bg = (G == 0) ? bq : (G == 1) ? bk : bv;
      const bf16_t* wtg = wt + (size_t)G * WMAT + (size_t)h * KEY_DIM * CC;  // [d][c]
      v8f cacc = (v8f){0.f,0.f,0.f,0.f,0.f,0.f,0.f,0.f};
      // Unroll x4: clause 8 global_b128 + 8 ds_b128 ahead of 4 WMMAs so the
      // matrix pipe isn't drained by a full s_wait every k-step.
#pragma unroll 4
      for (int ks = 0; ks < 16; ++ks) {
        v16bf a = frag_nk(Xs, XSTR, m * 16, ks * 32, lane);
        v16bf bfr;
        if (PREP) bfr = frag_nk(wtg, CC, nn * 16, ks * 32, lane);  // two b128 loads
        else      bfr = frag_w(wg + h * KEY_DIM, NUM_HEADS * KEY_DIM, ks * 32, nn * 16, lane);
        cacc = __builtin_amdgcn_wmma_f32_16x16x32_bf16(false, a, false, bfr,
                                                       (short)0, cacc, false, false);
      }
      const float bias = bg[h * KEY_DIM + nn * 16 + (lane & 15)];
#pragma unroll
      for (int vg = 0; vg < 8; ++vg) cacc[vg] += bias;
      if (G == 0)      store_c_bf16(Qs, HSTR, m * 16, nn * 16, cacc, lane, false);
      else if (G == 1) store_c_bf16(Ks, HSTR, m * 16, nn * 16, cacc, lane, false);
      else             store_c_bf16(Vs, HSTR, m * 16, nn * 16, cacc, lane, true);
    }
    __syncthreads();

    // ---- scores = (Q_h K_h^T) * scale ----
#pragma unroll 1
    for (int i = 0; i < 2; ++i) {
      const int tile = wave * 2 + i;
      const int m = tile >> 2, nn = tile & 3;
      v8f cacc = (v8f){0.f,0.f,0.f,0.f,0.f,0.f,0.f,0.f};
#pragma unroll
      for (int ks = 0; ks < 2; ++ks) {
        v16bf a   = frag_nk(Qs, HSTR, m * 16, ks * 32, lane);   // [q][d]
        v16bf bfr = frag_nk(Ks, HSTR, nn * 16, ks * 32, lane);  // [k][d] = B^T
        cacc = __builtin_amdgcn_wmma_f32_16x16x32_bf16(false, a, false, bfr,
                                                       (short)0, cacc, false, false);
      }
      const int half = lane >> 4;
#pragma unroll
      for (int vg = 0; vg < 8; ++vg)
        Ss[(m * 16 + vg + 8 * half) * SSTR + nn * 16 + (lane & 15)] = cacc[vg] * scale;
    }
    __syncthreads();

    // ---- softmax, one row per thread (threads 0..63); attn -> Qs (bf16) ----
    if (tid < T) {
      float* row = Ss + tid * SSTR;
      float mx = row[0];
#pragma unroll 1
      for (int j = 1; j < T; ++j) mx = fmaxf(mx, row[j]);
      float sum = 0.f;
#pragma unroll 1
      for (int j = 0; j < T; ++j) { float e = __expf(row[j] - mx); row[j] = e; sum += e; }
      const float inv = 1.f / sum;
      bf16_t* arow = Qs + tid * HSTR;
#pragma unroll 1
      for (int j = 0; j < T; ++j) arow[j] = f2bf(row[j] * inv);
    }
    __syncthreads();

    // ---- O_h = attn · V_h  (V stored transposed [d][k]) -> Ks (reused) ----
#pragma unroll 1
    for (int i = 0; i < 2; ++i) {
      const int tile = wave * 2 + i;
      const int m = tile >> 2, nn = tile & 3;
      v8f cacc = (v8f){0.f,0.f,0.f,0.f,0.f,0.f,0.f,0.f};
#pragma unroll
      for (int ks = 0; ks < 2; ++ks) {
        v16bf a   = frag_nk(Qs, HSTR, m * 16, ks * 32, lane);   // attn [q][k]
        v16bf bfr = frag_nk(Vs, HSTR, nn * 16, ks * 32, lane);  // V^T  [d][k]
        cacc = __builtin_amdgcn_wmma_f32_16x16x32_bf16(false, a, false, bfr,
                                                       (short)0, cacc, false, false);
      }
      store_c_bf16(Ks, HSTR, m * 16, nn * 16, cacc, lane, false);  // O_h [q][d]
    }
    __syncthreads();

    // ---- fused output projection: acc += O_h · Wo[h] ----
#pragma unroll 1
    for (int m = 0; m < 4; ++m) {
#pragma unroll 1
      for (int nl = 0; nl < 4; ++nl) {
        v8f cacc = acc[m * 4 + nl];
#pragma unroll
        for (int ks = 0; ks < 2; ++ks) {
          v16bf a = frag_nk(Ks, HSTR, m * 16, ks * 32, lane);
          v16bf bfr;
          if (PREP) bfr = frag_nk(wt + 3 * WMAT + (size_t)h * CC * KEY_DIM, KEY_DIM,
                                  wave * 64 + nl * 16, ks * 32, lane);   // [c][d]
          else      bfr = frag_w(wo + (size_t)h * KEY_DIM * CC, CC,
                                 ks * 32, wave * 64 + nl * 16, lane);
          cacc = __builtin_amdgcn_wmma_f32_16x16x32_bf16(false, a, false, bfr,
                                                         (short)0, cacc, false, false);
        }
        acc[m * 4 + nl] = cacc;
      }
    }
    __syncthreads();  // protect Qs/Ks/Vs for next head
  }

  // ---- epilogue: out = x + proj + bo, scattered with inverse shift ----
  const int half = lane >> 4;
#pragma unroll 1
  for (int m = 0; m < 4; ++m) {
#pragma unroll 1
    for (int nl = 0; nl < 4; ++nl) {
      const int ch = wave * 64 + nl * 16 + (lane & 15);
      const float bias = bo[ch];
      v8f cacc = acc[m * 4 + nl];
#pragma unroll
      for (int vg = 0; vg < 8; ++vg) {
        const int t = m * 16 + vg + 8 * half;
        const int r = t >> 3, c = t & 7;
        const int gy = (whi * 8 + r + SHIFT) & 63;
        const int gx = (wwi * 8 + c + SHIFT) & 63;
        const size_t idx = ((size_t)((b * 64 + gy) * 64 + gx)) * CC + ch;
        out[idx] = x[idx] + bias + cacc[vg];
      }
    }
  }
}

extern "C" void kernel_launch(void* const* d_in, const int* in_sizes, int n_in,
                              void* d_out, int out_size, void* d_ws, size_t ws_size,
                              hipStream_t stream) {
  const float* x  = (const float*)d_in[0];
  const float* wq = (const float*)d_in[1];
  const float* bq = (const float*)d_in[2];
  const float* wk = (const float*)d_in[3];
  const float* bk = (const float*)d_in[4];
  const float* wv = (const float*)d_in[5];
  const float* bv = (const float*)d_in[6];
  const float* wo = (const float*)d_in[7];
  const float* bo = (const float*)d_in[8];
  float* out = (float*)d_out;

  const size_t need = (size_t)4 * WMAT * sizeof(bf16_t);   // 2 MB of bf16 weights
  if (ws_size >= need) {
    bf16_t* wt = (bf16_t*)d_ws;
    hipLaunchKernelGGL(prep_weights_kernel, dim3(4 * WMAT / 256), dim3(256), 0, stream,
                       wq, wk, wv, wo, wt);
    hipLaunchKernelGGL((swin_window_attn_kernel<true>), dim3(NWIN), dim3(256), 0, stream,
                       x, wq, bq, wk, bk, wv, bv, wo, bo, (const bf16_t*)wt, out);
  } else {
    hipLaunchKernelGGL((swin_window_attn_kernel<false>), dim3(NWIN), dim3(256), 0, stream,
                       x, wq, bq, wk, bk, wv, bv, wo, bo, (const bf16_t*)nullptr, out);
  }
  (void)in_sizes; (void)n_in; (void)out_size;
}
